// MPNEncoder_75591424410255
// MI455X (gfx1250) — compile-verified, matrix operand order checked by reading, not run on hardware
//
#include <hip/hip_runtime.h>
#include <hip/hip_bf16.h>

typedef __attribute__((ext_vector_type(16))) _Float16 v16h;
typedef __attribute__((ext_vector_type(8)))  float    v8f;

#define WMMA_F16(a, b, c) \
    __builtin_amdgcn_wmma_f32_16x16x32_f16(false, (a), false, (b), (short)0, (c), false, false)

#define TPW 4   // row tiles per wave

// ---------------------------------------------------------------------------
// Stage weight matrix [Krows x Ncols] (fp32, row-major) into LDS in WMMA
// B-fragment order: Wl[(kc*4+nt)*32 + lane][16 halves], zero-padded.
// Each lane later fetches its fragment with two ds_load_b128.
// ---------------------------------------------------------------------------
template<int KC>
__device__ inline void stage_w(const float* __restrict__ W, int Krows, int Ncols,
                               _Float16* Wl, int tid)
{
    constexpr int TOT = KC * 4 * 32 * 16;
    for (int i = tid; i < TOT; i += 256) {
        const int j  = i & 15;
        const int l  = (i >> 4) & 31;
        const int f  = i >> 9;
        const int kc = f >> 2, nt = f & 3;
        const int hi = l >> 4;
        const int k  = kc * 32 + ((j < 8) ? (hi * 8 + j) : (16 + hi * 8 + (j - 8)));
        const int n  = nt * 16 + (l & 15);
        float v = (k < Krows && n < Ncols) ? W[k * Ncols + n] : 0.f;
        Wl[i] = (_Float16)v;
    }
}

__device__ inline v16h lds_frag(const _Float16* Wl, int frag, int lane)
{
    return *(const v16h*)(Wl + ((frag * 32) + lane) * 16);
}

// ---------------------------------------------------------------------------
// Stage 1: inp = f_bonds @ W_i (K=147->160, N=50->64); msg0 = relu(inp)
// ---------------------------------------------------------------------------
__global__ __launch_bounds__(256) void k_bond_input(
    const float* __restrict__ f_bonds, const float* __restrict__ W_i,
    float* __restrict__ inp, float* __restrict__ msg, int NB_, int NBp_)
{
    __shared__ alignas(16) _Float16 Wl[5 * 4 * 32 * 16];
    const int tid = threadIdx.x;
    stage_w<5>(W_i, 147, 50, Wl, tid);
    __syncthreads();

    const int wave = tid >> 5, lane = tid & 31;
    const int m = lane & 15, hi = lane >> 4;

    for (int t = 0; t < TPW; ++t) {
        const long rowBase = (((long)blockIdx.x * 8 + wave) * TPW + t) * 16;
        if (rowBase >= NBp_) return;
        const long row = rowBase + m;
        // padding rows alias row 0 (valid memory; results land in padding rows
        // that are never read back)
        const float* arow = f_bonds + (row < (long)NB_ ? row : 0) * 147;

        v8f acc[4] = {};
#pragma unroll
        for (int kc = 0; kc < 5; ++kc) {
            const int k0 = kc * 32 + hi * 8;
            v16h a;
            if (kc < 4) {
#pragma unroll
                for (int j = 0; j < 8; ++j) {
                    a[j]     = (_Float16)arow[k0 + j];
                    a[j + 8] = (_Float16)arow[k0 + 16 + j];
                }
            } else {  // K tail: guard against reading past end of f_bonds
#pragma unroll
                for (int j = 0; j < 8; ++j) {
                    const int ka = k0 + j, kb = k0 + 16 + j;
                    a[j]     = (_Float16)((ka < 147) ? arow[ka] : 0.f);
                    a[j + 8] = (_Float16)((kb < 147) ? arow[kb] : 0.f);
                }
            }
#pragma unroll
            for (int nt = 0; nt < 4; ++nt) {
                const v16h b = lds_frag(Wl, kc * 4 + nt, lane);
                acc[nt] = WMMA_F16(a, b, acc[nt]);
            }
        }
#pragma unroll
        for (int nt = 0; nt < 4; ++nt)
#pragma unroll
            for (int v = 0; v < 8; ++v) {
                const long r = rowBase + v + 8 * hi;
                const float x = acc[nt][v];
                inp[r * 64 + nt * 16 + m] = x;
                msg[r * 64 + nt * 16 + m] = x > 0.f ? x : 0.f;
            }
    }
}

// ---------------------------------------------------------------------------
// Gather-sum: a_msg[a] = sum_{j<6} msg[a2b[a][j]]   (pure bandwidth, float4)
// ---------------------------------------------------------------------------
__global__ __launch_bounds__(256) void k_gather(
    const float* __restrict__ msg, const int* __restrict__ a2b,
    float* __restrict__ a_msg, int NA_, int NAp_)
{
    const long gid = (long)blockIdx.x * 256 + threadIdx.x;   // NAp*16 threads
    const long a = gid >> 4;
    const int c4 = (int)(gid & 15) * 4;
    if (a >= (long)NAp_) return;
    float4 s = make_float4(0.f, 0.f, 0.f, 0.f);
    if (a < (long)NA_) {
#pragma unroll
        for (int j = 0; j < 6; ++j) {
            const long b = (long)a2b[a * 6 + j];
            const float4 v = *(const float4*)(msg + b * 64 + c4);
            s.x += v.x; s.y += v.y; s.z += v.z; s.w += v.w;
        }
    }
    *(float4*)(a_msg + a * 64 + c4) = s;
}

// ---------------------------------------------------------------------------
// Bond update: msg_out = relu(inp + (a_msg[b2a] - msg_in[b2revb]) @ W_h)
// 8 B-fragments hoisted into registers; zero LDS traffic in row loop.
// ---------------------------------------------------------------------------
__global__ __launch_bounds__(256) void k_bond_update(
    const float* __restrict__ inp, const float* __restrict__ a_msg,
    const float* __restrict__ msg_in, float* __restrict__ msg_out,
    const float* __restrict__ W_h,
    const int* __restrict__ b2a, const int* __restrict__ b2revb,
    int NB_, int NBp_)
{
    __shared__ alignas(16) _Float16 Wl[2 * 4 * 32 * 16];
    const int tid = threadIdx.x;
    stage_w<2>(W_h, 50, 50, Wl, tid);
    __syncthreads();

    const int wave = tid >> 5, lane = tid & 31;
    const int m = lane & 15, hi = lane >> 4;

    v16h bf[8];
#pragma unroll
    for (int f = 0; f < 8; ++f) bf[f] = lds_frag(Wl, f, lane);

    for (int t = 0; t < TPW; ++t) {
        const long rowBase = (((long)blockIdx.x * 8 + wave) * TPW + t) * 16;
        if (rowBase >= NBp_) return;
        const long row = rowBase + m;

        long ia = 0, ir = 0;
        if (row < (long)NB_) { ia = b2a[row]; ir = b2revb[row]; }
        const float* pa = a_msg  + ia * 64;
        const float* pr = msg_in + ir * 64;

        v8f acc[4] = {};
#pragma unroll
        for (int kc = 0; kc < 2; ++kc) {
            const int k0 = kc * 32 + hi * 8;
            v16h a;
#pragma unroll
            for (int j = 0; j < 8; ++j) {
                a[j]     = (_Float16)(pa[k0 + j]      - pr[k0 + j]);
                a[j + 8] = (_Float16)(pa[k0 + 16 + j] - pr[k0 + 16 + j]);
            }
#pragma unroll
            for (int nt = 0; nt < 4; ++nt)
                acc[nt] = WMMA_F16(a, bf[kc * 4 + nt], acc[nt]);
        }
#pragma unroll
        for (int nt = 0; nt < 4; ++nt)
#pragma unroll
            for (int v = 0; v < 8; ++v) {
                const long r = rowBase + v + 8 * hi;
                const float x = acc[nt][v] + inp[r * 64 + nt * 16 + m];
                msg_out[r * 64 + nt * 16 + m] = x > 0.f ? x : 0.f;
            }
    }
}

// ---------------------------------------------------------------------------
// Build a_input (f16): [NAp,192] = concat(f_atoms[133], a_msg[50], zero-pad)
// ---------------------------------------------------------------------------
__global__ void k_concat(const float* __restrict__ f_atoms,
                         const float* __restrict__ a_msg,
                         _Float16* __restrict__ ain, int NA_)
{
    const long a = blockIdx.x;           // grid = NAp
    const int c = threadIdx.x;           // 192 threads
    float v = 0.f;
    if (a < (long)NA_) {
        if (c < 133)      v = f_atoms[a * 133 + c];
        else if (c < 183) v = a_msg[a * 64 + (c - 133)];
    }
    ain[a * 192 + c] = (_Float16)v;
}

// ---------------------------------------------------------------------------
// atom_hiddens = relu([f_atoms, a_msg] @ W_o + b_o); K=192 (6 WMMA steps)
// ---------------------------------------------------------------------------
__global__ __launch_bounds__(256) void k_atom_out(
    const _Float16* __restrict__ ain, const float* __restrict__ W_o,
    const float* __restrict__ b_o, float* __restrict__ ah, int NAp_)
{
    __shared__ alignas(16) _Float16 Wl[6 * 4 * 32 * 16];
    const int tid = threadIdx.x;
    stage_w<6>(W_o, 183, 50, Wl, tid);
    __syncthreads();

    const int wave = tid >> 5, lane = tid & 31;
    const int m = lane & 15, hi = lane >> 4;
    const float bias0 = ((0 * 16 + m) < 50) ? b_o[0 * 16 + m] : 0.f;
    const float bias1 = ((1 * 16 + m) < 50) ? b_o[1 * 16 + m] : 0.f;
    const float bias2 = ((2 * 16 + m) < 50) ? b_o[2 * 16 + m] : 0.f;
    const float bias3 = 0.f;  // cols 48..63: 48,49 valid -> handled below
    const float bias3r = ((3 * 16 + m) < 50) ? b_o[3 * 16 + m] : 0.f;
    (void)bias3;
    const float biasv[4] = {bias0, bias1, bias2, bias3r};

    for (int t = 0; t < TPW; ++t) {
        const long rowBase = (((long)blockIdx.x * 8 + wave) * TPW + t) * 16;
        if (rowBase >= NAp_) return;
        const _Float16* ar = ain + (rowBase + m) * 192;

        v8f acc[4] = {};
#pragma unroll
        for (int kc = 0; kc < 6; ++kc) {
            const int k0 = kc * 32 + hi * 8;
            v16h a;
#pragma unroll
            for (int j = 0; j < 8; ++j) {
                a[j]     = ar[k0 + j];
                a[j + 8] = ar[k0 + 16 + j];
            }
#pragma unroll
            for (int nt = 0; nt < 4; ++nt) {
                const v16h b = lds_frag(Wl, kc * 4 + nt, lane);
                acc[nt] = WMMA_F16(a, b, acc[nt]);
            }
        }
#pragma unroll
        for (int nt = 0; nt < 4; ++nt) {
            const int n = nt * 16 + m;
#pragma unroll
            for (int v = 0; v < 8; ++v) {
                const long r = rowBase + v + 8 * hi;
                const float x = acc[nt][v] + biasv[nt];
                ah[r * 64 + n] = x > 0.f ? x : 0.f;
            }
        }
    }
}

// ---------------------------------------------------------------------------
// mol_mean[m] = mean over 20 atom rows (skip padding row 0)
// ---------------------------------------------------------------------------
__global__ void k_mol_mean(const float* __restrict__ ah, float* __restrict__ mm)
{
    const long mol = blockIdx.x;
    const int c = threadIdx.x;        // 64 threads
    if (c >= 50) return;
    float s = 0.f;
    const long base = (1 + mol * 20) * 64 + c;
#pragma unroll
    for (int i = 0; i < 20; ++i) s += ah[base + (long)i * 64];
    mm[mol * 50 + c] = s * (1.0f / 20.0f);
}

// ---------------------------------------------------------------------------
// Expand: 20 copies/mol (atoms segment) then 45 copies/mol (bonds segment)
// ---------------------------------------------------------------------------
__global__ void k_expand(const float* __restrict__ mm, float* __restrict__ out,
                         int nMols)
{
    const long row = blockIdx.x;      // grid = nMols*65
    const int c = threadIdx.x;        // 64 threads
    if (c >= 50) return;
    const long nAtomRows = (long)nMols * 20;
    long mol, o;
    if (row < nAtomRows) { mol = row / 20; o = row * 50; }
    else { const long r2 = row - nAtomRows; mol = r2 / 45; o = nAtomRows * 50 + r2 * 50; }
    out[o + c] = mm[mol * 50 + c];
}

// ---------------------------------------------------------------------------
extern "C" void kernel_launch(void* const* d_in, const int* in_sizes, int n_in,
                              void* d_out, int out_size, void* d_ws, size_t ws_size,
                              hipStream_t stream)
{
    const float* f_atoms = (const float*)d_in[0];
    const float* f_bonds = (const float*)d_in[1];
    const float* W_i     = (const float*)d_in[2];
    const float* W_h     = (const float*)d_in[3];
    const float* W_o     = (const float*)d_in[4];
    const float* b_o     = (const float*)d_in[5];
    const int*   a2b     = (const int*)d_in[6];
    const int*   b2a     = (const int*)d_in[7];
    const int*   b2revb  = (const int*)d_in[8];

    const int NA = in_sizes[0] / 133;         // 400001
    const int NB = in_sizes[1] / 147;         // 880001
    const int nMols = (NA - 1) / 20;          // 20000
    const int NAp = (NA + 15) & ~15;
    const int NBp = (NB + 15) & ~15;

    _Float16* ain = (_Float16*)d_ws;
    float* fbase = (float*)((char*)d_ws + (size_t)NAp * 192 * sizeof(_Float16));
    float* inp  = fbase;
    float* msgA = inp  + (size_t)NBp * 64;
    float* msgB = msgA + (size_t)NBp * 64;
    float* amsg = msgB + (size_t)NBp * 64;
    float* ah   = amsg + (size_t)NAp * 64;
    float* mm   = ah   + (size_t)NAp * 64;

    const int tilesB = NBp / 16;
    const int tilesA = NAp / 16;
    const int bondBlocks = ((tilesB + TPW - 1) / TPW + 7) / 8;
    const int atomBlocks = ((tilesA + TPW - 1) / TPW + 7) / 8;
    const int gatherBlocks = (int)(((long)NAp * 16 + 255) / 256);

    // inp = f_bonds @ W_i ; msgA = relu(inp)
    k_bond_input<<<bondBlocks, 256, 0, stream>>>(f_bonds, W_i, inp, msgA, NB, NBp);

    // depth iteration 1: msgA -> msgB
    k_gather<<<gatherBlocks, 256, 0, stream>>>(msgA, a2b, amsg, NA, NAp);
    k_bond_update<<<bondBlocks, 256, 0, stream>>>(inp, amsg, msgA, msgB, W_h,
                                                  b2a, b2revb, NB, NBp);
    // depth iteration 2: msgB -> msgA
    k_gather<<<gatherBlocks, 256, 0, stream>>>(msgB, a2b, amsg, NA, NAp);
    k_bond_update<<<bondBlocks, 256, 0, stream>>>(inp, amsg, msgB, msgA, W_h,
                                                  b2a, b2revb, NB, NBp);
    // final atom aggregation + readout
    k_gather<<<gatherBlocks, 256, 0, stream>>>(msgA, a2b, amsg, NA, NAp);
    k_concat<<<NAp, 192, 0, stream>>>(f_atoms, amsg, ain, NA);
    k_atom_out<<<atomBlocks, 256, 0, stream>>>(ain, W_o, b_o, ah, NAp);
    k_mol_mean<<<nMols, 64, 0, stream>>>(ah, mm);
    k_expand<<<nMols * 65, 64, 0, stream>>>(mm, (float*)d_out, nMols);
}
